// RoPEAttention_58712202936757
// MI455X (gfx1250) — matrix operand order repaired and checked by reading, not compiled, not run on hardware
//
#include <hip/hip_runtime.h>
#include <cstdint>

// ---------------------------------------------------------------------------
// RoPE-2D attention, MI455X (gfx1250, wave32, WMMA, async LDS copies).
// fp32 inputs -> one cheap bf16 pre-convert pass (~3us of HBM time), then all
// GEMM stages run v_wmma_f32_16x16x32_bf16 with double-buffered
// GLOBAL_LOAD_ASYNC_TO_LDS_B128 tile staging (ASYNCcnt pipeline).
// ---------------------------------------------------------------------------

#define NHEAD    16
#define HEAD_DIM 64
#define GRIDSZ   32
#define SEQ      (GRIDSZ * GRIDSZ)      // 1024
#define BATCH    8
#define DMODEL   (NHEAD * HEAD_DIM)     // 1024
#define MROWS    (BATCH * SEQ)          // 8192

typedef __attribute__((ext_vector_type(16))) __bf16        v16bf;
typedef __attribute__((ext_vector_type(2)))  __bf16        v2bf;
typedef __attribute__((ext_vector_type(8)))  float         v8f;
typedef __attribute__((ext_vector_type(4)))  float         v4f;
typedef __attribute__((ext_vector_type(4)))  int           v4i;
typedef __attribute__((ext_vector_type(4)))  unsigned int  v4u;
typedef __attribute__((ext_vector_type(2)))  unsigned int  v2u;

union FragU { v4u u[2]; v16bf v; };
union FragW { unsigned w[8]; v16bf v; };

// ---- bf16 conversion via hardware converters ------------------------------
__device__ __forceinline__ unsigned pack_bf16(float a, float b) {
#if defined(__has_builtin) && __has_builtin(__builtin_amdgcn_cvt_pk_bf16_f32)
    v2bf t = __builtin_amdgcn_cvt_pk_bf16_f32(a, b);
    return __builtin_bit_cast(unsigned, t);
#else
    __bf16 x = (__bf16)a, y = (__bf16)b;
    return (unsigned)__builtin_bit_cast(unsigned short, x) |
           ((unsigned)__builtin_bit_cast(unsigned short, y) << 16);
#endif
}
__device__ __forceinline__ unsigned short f2bf_s(float f) {
    __bf16 h = (__bf16)f;
    return __builtin_bit_cast(unsigned short, h);
}

// ---- async global->LDS copy (CDNA5), with synchronous fallback ------------
#if defined(__has_builtin)
#if __has_builtin(__builtin_amdgcn_global_load_async_to_lds_b128)
#define HAVE_ASYNC_LDS 1
#endif
#endif

#ifdef HAVE_ASYNC_LDS
// Signature (from hipcc diagnostic): (v4i addrspace(1)*, v4i addrspace(3)*, Ii, Ii)
typedef __attribute__((address_space(1))) v4i as1_v4i;
typedef __attribute__((address_space(3))) v4i as3_v4i;
__device__ __forceinline__ void async_b128(const void* g, void* l) {
    as1_v4i* gp = (as1_v4i*)(unsigned long long)(uintptr_t)g;
    as3_v4i* lp = (as3_v4i*)(unsigned)(uintptr_t)l;   // low 32b = LDS offset
    __builtin_amdgcn_global_load_async_to_lds_b128(gp, lp, 0, 0);
}
#if __has_builtin(__builtin_amdgcn_s_wait_asynccnt)
#define ASYNC_WAIT(n) __builtin_amdgcn_s_wait_asynccnt(n)
#else
#define ASYNC_WAIT(n)                                                         \
    do {                                                                      \
        if ((n) == 0)      asm volatile("s_wait_asynccnt 0x0" ::: "memory");  \
        else if ((n) == 2) asm volatile("s_wait_asynccnt 0x2" ::: "memory");  \
        else               asm volatile("s_wait_asynccnt 0x4" ::: "memory");  \
    } while (0)
#endif
#else
__device__ __forceinline__ void async_b128(const void* g, void* l) {
    *(v4u*)l = *(const v4u*)g;          // fallback: regular copy through VGPRs
}
#define ASYNC_WAIT(n) do {} while (0)
#endif

__device__ __forceinline__ v8f zero8() {
    v8f z = {0.f, 0.f, 0.f, 0.f, 0.f, 0.f, 0.f, 0.f};
    return z;
}

__device__ __forceinline__ v8f wmma_bf16(v16bf a, v16bf b, v8f c) {
    return __builtin_amdgcn_wmma_f32_16x16x32_bf16(
        false, a, false, b, (short)0, c, false, false);
}

// A-frag (16x32 MxK): lane row M=lane%16; k = {c..c+7, c+16..c+23}, c=8*(lane>=16)
__device__ __forceinline__ v16bf load_frag_a(const unsigned short* buf,
                                             int row, int ldk, int koff, int lhi) {
    const unsigned short* p = buf + row * ldk + koff + 8 * lhi;
    FragU f;
    f.u[0] = *(const v4u*)(p);
    f.u[1] = *(const v4u*)(p + 16);
    return f.v;
}
// B-frag (32x16 KxN): lane col N=lane%16; k = {c..c+7, c+16..c+23}, c=8*(lane>=16). buf=[N][K].
__device__ __forceinline__ v16bf load_frag_b(const unsigned short* buf,
                                             int row, int ldk, int koff, int lhi) {
    const unsigned short* p = buf + row * ldk + koff + 8 * lhi;
    FragU f;
    f.u[0] = *(const v4u*)(p);
    f.u[1] = *(const v4u*)(p + 16);
    return f.v;
}

// ---------------------------------------------------------------------------
// Kernel 0: fp32 -> bf16 bulk convert (x, qkv_w, out_w). Pure bandwidth.
// ---------------------------------------------------------------------------
__global__ __launch_bounds__(256) void to_bf16(const float* __restrict__ src,
                                               unsigned short* __restrict__ dst,
                                               int n4) {
    int i = blockIdx.x * 256 + threadIdx.x;
    if (i < n4) {
        v4f a = *(const v4f*)(src + (size_t)i * 4);
        v2u p;
        p.x = pack_bf16(a.x, a.y);
        p.y = pack_bf16(a.z, a.w);
        *(v2u*)(dst + (size_t)i * 4) = p;
    }
}

// ---------------------------------------------------------------------------
// Kernel 1: QKV = x @ qkv_w.T + b with fused RoPE-2D. bf16 in, bf16 q/k/v out
// [B,H,N,64]. 128x128 block, K-step 32, double-buffered async LDS staging.
// Each wave owns 64 aligned N-cols = one (s,head): RoPE partner dh^16 sits in
// the adjacent accumulator tile of the same wave/lane/slot.
// ---------------------------------------------------------------------------
__global__ __launch_bounds__(256) void qkv_rope_gemm(
    const unsigned short* __restrict__ xbf,   // [8192][1024] bf16
    const unsigned short* __restrict__ wbf,   // [3072][1024] bf16
    const float* __restrict__ bias,           // [3072]
    unsigned short* __restrict__ qbf,
    unsigned short* __restrict__ kbf,
    unsigned short* __restrict__ vbf)
{
    __shared__ unsigned short As[2][128 * 32];
    __shared__ unsigned short Bs[2][128 * 32];

    const int tid  = threadIdx.x;
    const int lane = tid & 31, wave = tid >> 5;
    const int lhi  = lane >> 4, lmod = lane & 15;
    const int wm   = wave >> 1, wn = wave & 1;
    const int m0   = blockIdx.x * 128;
    const int n0   = blockIdx.y * 128;

    auto stage = [&](int buf, int kt) {
#pragma unroll
        for (int i = 0; i < 2; i++) {
            int f = tid + 256 * i, row = f >> 2, c8 = (f & 3) * 8;
            async_b128(xbf + (size_t)(m0 + row) * DMODEL + kt + c8,
                       &As[buf][row * 32 + c8]);
            async_b128(wbf + (size_t)(n0 + row) * DMODEL + kt + c8,
                       &Bs[buf][row * 32 + c8]);
        }
    };

    v8f acc[2][4];
#pragma unroll
    for (int mi = 0; mi < 2; mi++)
#pragma unroll
        for (int ni = 0; ni < 4; ni++) acc[mi][ni] = zero8();

    stage(0, 0);
    int buf = 0;
    for (int kt = 0; kt < DMODEL; kt += 32) {
        __syncthreads();                              // prev compute done
        if (kt + 32 < DMODEL) { stage(buf ^ 1, kt + 32); ASYNC_WAIT(4); }
        else                  { ASYNC_WAIT(0); }
        __syncthreads();                              // cur tile visible

        v16bf af[2], bfr[4];
#pragma unroll
        for (int mi = 0; mi < 2; mi++)
            af[mi] = load_frag_a(&As[buf][0], 32 * wm + 16 * mi + lmod, 32, 0, lhi);
#pragma unroll
        for (int ni = 0; ni < 4; ni++)
            bfr[ni] = load_frag_b(&Bs[buf][0], 64 * wn + 16 * ni + lmod, 32, 0, lhi);
#pragma unroll
        for (int ni = 0; ni < 4; ni++)
#pragma unroll
            for (int mi = 0; mi < 2; mi++)
                acc[mi][ni] = wmma_bf16(af[mi], bfr[ni], acc[mi][ni]);
        buf ^= 1;
    }

    // Epilogue: bias + RoPE-2D -> bf16 [B,H,N,64]
    const int cg = n0 + 64 * wn;
    const int s  = cg >> 10;                 // 0=q 1=k 2=v
    const int h  = (cg & 1023) >> 6;
    unsigned short* dst = (s == 0) ? qbf : (s == 1) ? kbf : vbf;

    float bias4[4];
#pragma unroll
    for (int ni = 0; ni < 4; ni++) bias4[ni] = bias[cg + 16 * ni + lmod];
    const float invf = __expf(-0.5756462732485114f * (float)lmod); // 1e4^{-lmod/16}

#pragma unroll
    for (int mi = 0; mi < 2; mi++) {
        float vals[4][8];
#pragma unroll
        for (int ni = 0; ni < 4; ni++)
#pragma unroll
            for (int r = 0; r < 8; r++) vals[ni][r] = acc[mi][ni][r] + bias4[ni];
#pragma unroll
        for (int ni = 0; ni < 4; ni++) {
            const int dh = 16 * ni + lmod;
#pragma unroll
            for (int r = 0; r < 8; r++) {
                int m = m0 + 32 * wm + 16 * mi + r + 8 * lhi;
                int b = m >> 10, nt = m & 1023;
                float o;
                if (s < 2) {
                    int pos   = (ni < 2) ? (nt >> 5) : (nt & 31); // y vs x part
                    float ang = (float)pos * invf, sv, cv;
                    __sincosf(ang, &sv, &cv);
                    float rot = (ni & 1) ? vals[ni ^ 1][r] : -vals[ni ^ 1][r];
                    o = vals[ni][r] * cv + rot * sv;
                } else {
                    o = vals[ni][r];
                }
                size_t a = (((size_t)b * NHEAD + h) * SEQ + nt) * HEAD_DIM + dh;
                dst[a] = f2bf_s(o);
            }
        }
    }
}

// ---------------------------------------------------------------------------
// Kernel 2: attention per (b,h, 64-query block). 64x1024 fp32 score slab in
// LDS (256KB of 320KB) -> plain two-pass softmax. K tiles streamed through a
// double-buffered async pipeline.
// ---------------------------------------------------------------------------
#define ATT_SMEM (64 * 1024 * 4 + 64 * 64 * 2 + 2 * 64 * 64 * 2)  // 286720

__global__ __launch_bounds__(256) void rope_attn(
    const unsigned short* __restrict__ qbf,
    const unsigned short* __restrict__ kbf,
    const unsigned short* __restrict__ vbf,
    unsigned short* __restrict__ obf)      // [B,N,H,64] bf16
{
    extern __shared__ char smem[];
    float*          sc  = (float*)smem;                            // [64][1024]
    unsigned short* qt  = (unsigned short*)(smem + 64 * 1024 * 4); // [64][64]
    unsigned short* kvb = qt + 64 * 64;                            // [2][64][64]
    float*          red = (float*)qt;                              // post-phase-1

    const int tid  = threadIdx.x;
    const int lane = tid & 31, wave = tid >> 5;
    const int lhi  = lane >> 4, lmod = lane & 15;
    const int mi   = wave & 3;           // 16 q-rows
    const int np   = wave >> 2;          // 32-col half
    const int qb   = blockIdx.x & 15;
    const int bh   = blockIdx.x >> 4;
    const size_t base = (size_t)bh * SEQ * HEAD_DIM;

    auto stageK = [&](int buf, int kb) {
#pragma unroll
        for (int i = 0; i < 2; i++) {
            int l8 = tid * 2 + i, r = l8 >> 3, c8 = (l8 & 7) * 8;
            async_b128(kbf + base + (size_t)(kb * 64 + r) * 64 + c8,
                       &kvb[buf * 4096 + r * 64 + c8]);
        }
    };

    // stage Q (async) + first K tile
#pragma unroll
    for (int i = 0; i < 2; i++) {
        int l8 = tid * 2 + i, r = l8 >> 3, c8 = (l8 & 7) * 8;
        async_b128(qbf + base + (size_t)(qb * 64 + r) * 64 + c8, &qt[r * 64 + c8]);
    }
    stageK(0, 0);

    // -------- Phase 1: S = scale * Q @ K^T ---------------------------------
    int buf = 0;
    for (int kb = 0; kb < 16; kb++) {
        __syncthreads();
        if (kb + 1 < 16) { stageK(buf ^ 1, kb + 1); ASYNC_WAIT(2); }
        else             { ASYNC_WAIT(0); }
        __syncthreads();
        const unsigned short* kt_ = &kvb[buf * 4096];

        v8f s0 = zero8(), s1 = zero8();
#pragma unroll
        for (int d0 = 0; d0 < 64; d0 += 32) {
            v16bf fa = load_frag_a(qt, 16 * mi + lmod, 64, d0, lhi);
            v16bf b0 = load_frag_b(kt_, 32 * np + lmod,      64, d0, lhi);
            v16bf b1 = load_frag_b(kt_, 32 * np + 16 + lmod, 64, d0, lhi);
            s0 = wmma_bf16(fa, b0, s0);
            s1 = wmma_bf16(fa, b1, s1);
        }
        const float scale = 0.125f;       // 64^-0.5
#pragma unroll
        for (int r = 0; r < 8; r++) {
            int q = 16 * mi + r + 8 * lhi;
            sc[q * 1024 + kb * 64 + 32 * np + lmod]      = s0[r] * scale;
            sc[q * 1024 + kb * 64 + 32 * np + 16 + lmod] = s1[r] * scale;
        }
        buf ^= 1;
    }
    __syncthreads();

    // -------- Phase 2: softmax, 4 threads per 1024-wide row ----------------
    {
        int r = tid >> 2, cq = tid & 3;
        float* row = sc + r * 1024 + cq * 256;
        float mx = -3.0e38f;
        for (int j = 0; j < 256; j++) mx = fmaxf(mx, row[j]);
        red[r * 4 + cq] = mx;
        __syncthreads();
        float rm = fmaxf(fmaxf(red[r * 4 + 0], red[r * 4 + 1]),
                         fmaxf(red[r * 4 + 2], red[r * 4 + 3]));
        __syncthreads();
        float sum = 0.f;
        for (int j = 0; j < 256; j++) {
            float e = __expf(row[j] - rm);
            row[j] = e;
            sum += e;
        }
        red[r * 4 + cq] = sum;
        __syncthreads();
        float inv = 1.0f / (red[r * 4 + 0] + red[r * 4 + 1] +
                            red[r * 4 + 2] + red[r * 4 + 3]);
        for (int j = 0; j < 256; j++) row[j] *= inv;
    }
    __syncthreads();

    // -------- Phase 3: O = P @ V (P fp32 -> bf16 frags; V transposed) ------
    v8f o0 = zero8(), o1 = zero8();
    for (int kb = 0; kb < 16; kb++) {
        __syncthreads();
#pragma unroll
        for (int i = 0; i < 2; i++) {                 // stage V^T [d][kk]
            int l8 = tid * 2 + i;
            int kk = l8 >> 3, dh0 = (l8 & 7) * 8;
            v4u vv = *(const v4u*)(vbf + base + (size_t)(kb * 64 + kk) * 64 + dh0);
            unsigned int wrd[4] = {vv.x, vv.y, vv.z, vv.w};
#pragma unroll
            for (int j = 0; j < 8; j++)
                kvb[(dh0 + j) * 64 + kk] =
                    (unsigned short)(wrd[j >> 1] >> ((j & 1) * 16));
        }
        __syncthreads();

#pragma unroll
        for (int kk0 = 0; kk0 < 64; kk0 += 32) {
            const float* prow =
                sc + (16 * mi + lmod) * 1024 + kb * 64 + kk0 + 8 * lhi;
            v4f p0 = *(const v4f*)(prow);
            v4f p1 = *(const v4f*)(prow + 4);
            v4f p2 = *(const v4f*)(prow + 16);
            v4f p3 = *(const v4f*)(prow + 20);
            FragW fw;
            fw.w[0] = pack_bf16(p0.x, p0.y); fw.w[1] = pack_bf16(p0.z, p0.w);
            fw.w[2] = pack_bf16(p1.x, p1.y); fw.w[3] = pack_bf16(p1.z, p1.w);
            fw.w[4] = pack_bf16(p2.x, p2.y); fw.w[5] = pack_bf16(p2.z, p2.w);
            fw.w[6] = pack_bf16(p3.x, p3.y); fw.w[7] = pack_bf16(p3.z, p3.w);

            v16bf b0 = load_frag_b(kvb, 32 * np + lmod,      64, kk0, lhi);
            v16bf b1 = load_frag_b(kvb, 32 * np + 16 + lmod, 64, kk0, lhi);
            o0 = wmma_bf16(fw.v, b0, o0);
            o1 = wmma_bf16(fw.v, b1, o1);
        }
    }

    const int b = bh >> 4, h = bh & 15;
#pragma unroll
    for (int r = 0; r < 8; r++) {
        int qg = qb * 64 + 16 * mi + r + 8 * lhi;
        size_t a0 = (((size_t)b * SEQ + qg) * NHEAD + h) * HEAD_DIM;
        obf[a0 + 32 * np + lmod]      = f2bf_s(o0[r]);
        obf[a0 + 32 * np + 16 + lmod] = f2bf_s(o1[r]);
    }
}

// ---------------------------------------------------------------------------
// Kernel 3: out = O @ out_w.T + out_b (bf16 x bf16 -> fp32). Async staging.
// ---------------------------------------------------------------------------
__global__ __launch_bounds__(256) void out_proj_gemm(
    const unsigned short* __restrict__ a,    // [8192][1024] bf16
    const unsigned short* __restrict__ wbf,  // [1024][1024] bf16
    const float* __restrict__ bias,
    float* __restrict__ out)
{
    __shared__ unsigned short As[2][128 * 32];
    __shared__ unsigned short Bs[2][128 * 32];

    const int tid  = threadIdx.x;
    const int lane = tid & 31, wave = tid >> 5;
    const int lhi  = lane >> 4, lmod = lane & 15;
    const int wm   = wave >> 1, wn = wave & 1;
    const int m0   = blockIdx.x * 128;
    const int n0   = blockIdx.y * 128;

    auto stage = [&](int buf, int kt) {
#pragma unroll
        for (int i = 0; i < 2; i++) {
            int f = tid + 256 * i, row = f >> 2, c8 = (f & 3) * 8;
            async_b128(a   + (size_t)(m0 + row) * DMODEL + kt + c8,
                       &As[buf][row * 32 + c8]);
            async_b128(wbf + (size_t)(n0 + row) * DMODEL + kt + c8,
                       &Bs[buf][row * 32 + c8]);
        }
    };

    v8f acc[2][4];
#pragma unroll
    for (int mi = 0; mi < 2; mi++)
#pragma unroll
        for (int ni = 0; ni < 4; ni++) acc[mi][ni] = zero8();

    stage(0, 0);
    int buf = 0;
    for (int kt = 0; kt < DMODEL; kt += 32) {
        __syncthreads();
        if (kt + 32 < DMODEL) { stage(buf ^ 1, kt + 32); ASYNC_WAIT(4); }
        else                  { ASYNC_WAIT(0); }
        __syncthreads();

        v16bf af[2], bfr[4];
#pragma unroll
        for (int mi = 0; mi < 2; mi++)
            af[mi] = load_frag_a(&As[buf][0], 32 * wm + 16 * mi + lmod, 32, 0, lhi);
#pragma unroll
        for (int ni = 0; ni < 4; ni++)
            bfr[ni] = load_frag_b(&Bs[buf][0], 64 * wn + 16 * ni + lmod, 32, 0, lhi);
#pragma unroll
        for (int ni = 0; ni < 4; ni++)
#pragma unroll
            for (int mi = 0; mi < 2; mi++)
                acc[mi][ni] = wmma_bf16(af[mi], bfr[ni], acc[mi][ni]);
        buf ^= 1;
    }

#pragma unroll
    for (int mi = 0; mi < 2; mi++)
#pragma unroll
        for (int ni = 0; ni < 4; ni++) {
            int col = n0 + 64 * wn + 16 * ni + lmod;
            float bb = bias[col];
#pragma unroll
            for (int r = 0; r < 8; r++) {
                int row = m0 + 32 * wm + 16 * mi + r + 8 * lhi;
                out[(size_t)row * DMODEL + col] = acc[mi][ni][r] + bb;
            }
        }
}

// ---------------------------------------------------------------------------
extern "C" void kernel_launch(void* const* d_in, const int* in_sizes, int n_in,
                              void* d_out, int out_size, void* d_ws, size_t ws_size,
                              hipStream_t stream)
{
    const float* x     = (const float*)d_in[0];
    const float* qkv_w = (const float*)d_in[1];
    const float* qkv_b = (const float*)d_in[2];
    const float* out_w = (const float*)d_in[3];
    const float* out_b = (const float*)d_in[4];
    float* out = (float*)d_out;

    const size_t TEN = (size_t)BATCH * NHEAD * SEQ * HEAD_DIM;   // 8,388,608
    unsigned short* qbf    = (unsigned short*)d_ws;
    unsigned short* kbf    = qbf + TEN;
    unsigned short* vbf    = kbf + TEN;
    unsigned short* obf    = vbf + TEN;
    unsigned short* xbf    = obf + TEN;                          // [8192][1024]
    unsigned short* wqkvbf = xbf + (size_t)MROWS * DMODEL;       // [3072][1024]
    unsigned short* woutbf = wqkvbf + (size_t)3 * DMODEL * DMODEL; // [1024][1024]

    (void)hipFuncSetAttribute(reinterpret_cast<const void*>(&rope_attn),
                              hipFuncAttributeMaxDynamicSharedMemorySize,
                              ATT_SMEM);

    // bf16 pre-convert (pure-bandwidth, ~70MB total traffic)
    to_bf16<<<(MROWS * DMODEL / 4 + 255) / 256, 256, 0, stream>>>(
        x, xbf, MROWS * DMODEL / 4);
    to_bf16<<<(3 * DMODEL * DMODEL / 4 + 255) / 256, 256, 0, stream>>>(
        qkv_w, wqkvbf, 3 * DMODEL * DMODEL / 4);
    to_bf16<<<(DMODEL * DMODEL / 4 + 255) / 256, 256, 0, stream>>>(
        out_w, woutbf, DMODEL * DMODEL / 4);

    qkv_rope_gemm<<<dim3(MROWS / 128, (3 * DMODEL) / 128), 256, 0, stream>>>(
        xbf, wqkvbf, qkv_b, qbf, kbf, vbf);

    rope_attn<<<BATCH * NHEAD * (SEQ / 64), 256, ATT_SMEM, stream>>>(
        qbf, kbf, vbf, obf);

    out_proj_gemm<<<dim3(MROWS / 128, DMODEL / 128), 256, 0, stream>>>(
        obf, woutbf, out_b, out);
}
